// EPD_with_sampling_44427141710345
// MI455X (gfx1250) — compile-verified, matrix operand order checked by reading, not run on hardware
//
#include <hip/hip_runtime.h>
#include <hip/hip_bf16.h>

typedef __attribute__((ext_vector_type(16))) _Float16 v16h;
typedef __attribute__((ext_vector_type(8)))  _Float16 v8h;
typedef __attribute__((ext_vector_type(4)))  _Float16 v4h;
typedef __attribute__((ext_vector_type(8)))  float    v8f;
typedef __attribute__((ext_vector_type(4)))  float    v4f;

#define DEV static __device__ __forceinline__

constexpr int N_   = 50000;
constexpr int E_   = 500000;
constexpr int G_   = 8;
constexpr int H_   = 64;
constexpr int OUT_ = 4;
constexpr int NT   = N_ / 16;   // 3125 node tiles (exact)
constexpr int ET   = E_ / 16;   // 31250 edge tiles (exact)

// ---------------------------------------------------------------------------
// Fragment builders per CDNA5 ISA 7.12.2.
// 16-bit A (16x32): lane=M (lane&15), K runs kb..kb+7 and kb+16..kb+23 where
// kb = kbase + (lane&16 ? 8 : 0)  ->  two contiguous 8-half (16B) LDS loads.
// B is pre-swizzled into exact fragment order at weight-staging time, so a
// B-fragment is one contiguous 32B LDS read per lane.
// C/D: VGPR r -> M = r + (lane&16 ? 8 : 0), N = lane&15 (+16*t).
// ---------------------------------------------------------------------------
DEV v16h frag_A(const _Float16* a, int lda, int kbase, int lane) {
  const int m  = lane & 15;
  const int kb = kbase + ((lane & 16) ? 8 : 0);
  const _Float16* p = a + m * lda + kb;
  v8h lo = *(const v8h*)(p);
  v8h hi = *(const v8h*)(p + 16);
  return __builtin_shufflevector(lo, hi, 0, 1, 2, 3, 4, 5, 6, 7,
                                 8, 9, 10, 11, 12, 13, 14, 15);
}

DEV v16h frag_Bsw(const _Float16* wsw, int kc, int t, int ntiles, int lane) {
  const _Float16* p = wsw + (((kc * ntiles + t) * 32 + lane) << 4);
  v8h lo = *(const v8h*)(p);
  v8h hi = *(const v8h*)(p + 8);
  return __builtin_shufflevector(lo, hi, 0, 1, 2, 3, 4, 5, 6, 7,
                                 8, 9, 10, 11, 12, 13, 14, 15);
}

DEV v8f wmma_f16(v16h a, v16h b, v8f c) {
  return __builtin_amdgcn_wmma_f32_16x16x32_f16(false, a, false, b,
                                                (short)0, c, false, false);
}

DEV v4h cvt4(v4f v) {
  v4h r;
#pragma unroll
  for (int j = 0; j < 4; ++j) r[j] = (_Float16)v[j];
  return r;
}

// Stage a Ksrc x ncolsrc fp32 weight matrix into f16 *fragment-swizzled*
// LDS layout: dst[(((kc*ntiles)+t)*32 + lane)*16 + j] = W[k(kc,lane,j)][n(t,lane)]
// (zero-padded beyond Ksrc/ncolsrc).
DEV void stage_Bsw(_Float16* dst, const float* src, int Ksrc, int ncolsrc,
                   int KCH, int ntiles, int tid, int nth) {
  const int total = KCH * ntiles * 512;
  for (int i = tid; i < total; i += nth) {
    int j    = i & 15;
    int lane = (i >> 4) & 31;
    int rest = i >> 9;               // kc*ntiles + t
    int t    = rest % ntiles;
    int kc   = rest / ntiles;
    int k = kc * 32 + ((lane & 16) ? 8 : 0) + j + ((j >= 8) ? 8 : 0);
    int n = t * 16 + (lane & 15);
    float v = (k < Ksrc && n < ncolsrc) ? src[k * ncolsrc + n] : 0.0f;
    dst[i] = (_Float16)v;
  }
}

// Two-layer wave MLP on a 16-row tile: out = relu(in*W1 + b1)*W2 + b2.
// hidT may alias inT (layer-1 fully consumes inT before hid is written;
// per-wave LDS ops are in-order).
template <int KPAD, int KC1, int NT2>
DEV void wave_mlp2(const _Float16* inT,
                   const _Float16* __restrict__ W1sw, const float* __restrict__ b1,
                   const _Float16* __restrict__ W2sw, const float* __restrict__ b2,
                   _Float16* hidT, int lane, v8f out[NT2]) {
  v8f a1[4] = {};
#pragma unroll
  for (int kc = 0; kc < KC1; ++kc) {
    v16h af = frag_A(inT, KPAD, kc * 32, lane);
#pragma unroll
    for (int t = 0; t < 4; ++t)
      a1[t] = wmma_f16(af, frag_Bsw(W1sw, kc, t, 4, lane), a1[t]);
  }
  const int n0 = lane & 15;
  const int mh = (lane & 16) ? 8 : 0;
#pragma unroll
  for (int t = 0; t < 4; ++t) {
    const float bb = b1[t * 16 + n0];
#pragma unroll
    for (int r = 0; r < 8; ++r)
      hidT[(mh + r) * 64 + t * 16 + n0] = (_Float16)fmaxf(a1[t][r] + bb, 0.0f);
  }
  v8f a2[NT2] = {};
#pragma unroll
  for (int kc = 0; kc < 2; ++kc) {
    v16h af = frag_A(hidT, 64, kc * 32, lane);
#pragma unroll
    for (int t = 0; t < NT2; ++t)
      a2[t] = wmma_f16(af, frag_Bsw(W2sw, kc, t, NT2, lane), a2[t]);
  }
#pragma unroll
  for (int t = 0; t < NT2; ++t) {
    const float bb = b2[t * 16 + n0];
#pragma unroll
    for (int r = 0; r < 8; ++r) a2[t][r] += bb;
    out[t] = a2[t];
  }
}

// ---------------------------------------------------------------------------
// Encoder: h = MLP 8->64->64 (concat(x, x_mask)), K padded to 32.
// ---------------------------------------------------------------------------
__global__ __launch_bounds__(128) void k_encode(
    const float* __restrict__ x, const float* __restrict__ xmask,
    const float* __restrict__ w1, const float* __restrict__ b1,
    const float* __restrict__ w2, const float* __restrict__ b2,
    float* __restrict__ h) {
  __shared__ alignas(16) _Float16 sW1[1 * 4 * 512];
  __shared__ alignas(16) _Float16 sW2[2 * 4 * 512];
  __shared__ float sB1[64], sB2[64];
  __shared__ alignas(16) _Float16 sIn[4][16 * 32];
  __shared__ alignas(16) _Float16 sHid[4][16 * 64];
  const int tid = threadIdx.x, nth = blockDim.x;
  stage_Bsw(sW1, w1, 8, 64, 1, 4, tid, nth);
  stage_Bsw(sW2, w2, 64, 64, 2, 4, tid, nth);
  for (int i = tid; i < 64; i += nth) { sB1[i] = b1[i]; sB2[i] = b2[i]; }
  __syncthreads();
  const int wave = tid >> 5, lane = tid & 31;
  const int tile = blockIdx.x * 4 + wave;
  if (tile >= NT) return;
  const int row0 = tile * 16;
  for (int i = lane; i < 16 * 8; i += 32) {   // 8 groups of 4 per row
    int r = i >> 3, g = i & 7;
    int node = row0 + r;
    v4f v = {};
    if (g == 0) {
      v[0] = x[node * 5 + 0]; v[1] = x[node * 5 + 1];
      v[2] = x[node * 5 + 2]; v[3] = x[node * 5 + 3];
    } else if (g == 1) {
      v[0] = x[node * 5 + 4];
      v[1] = xmask[node * 3 + 0]; v[2] = xmask[node * 3 + 1];
      v[3] = xmask[node * 3 + 2];
    }
    *(v4h*)(sIn[wave] + r * 32 + g * 4) = cvt4(v);
  }
  v8f acc[4];
  wave_mlp2<32, 1, 4>(sIn[wave], sW1, sB1, sW2, sB2, sHid[wave], lane, acc);
  const int n0 = lane & 15, mh = (lane & 16) ? 8 : 0;
#pragma unroll
  for (int t = 0; t < 4; ++t)
#pragma unroll
    for (int r = 0; r < 8; ++r)
      h[(size_t)(row0 + mh + r) * 64 + t * 16 + n0] = acc[t][r];
}

// ---------------------------------------------------------------------------
// Per-graph stats (hierarchical LDS -> global atomics).
// ---------------------------------------------------------------------------
__global__ __launch_bounds__(256) void k_stats(
    const float* __restrict__ h, const int* __restrict__ batch,
    const float* __restrict__ xmask, int with_bc,
    float* __restrict__ gsum, float* __restrict__ gcnt,
    float* __restrict__ bcsum, float* __restrict__ bccnt) {
  __shared__ float sg[G_ * H_], sb[G_ * H_], sc[G_], sbc[G_];
  const int tid = threadIdx.x;
  for (int i = tid; i < G_ * H_; i += 256) { sg[i] = 0.f; sb[i] = 0.f; }
  if (tid < G_) { sc[tid] = 0.f; sbc[tid] = 0.f; }
  __syncthreads();
  const int c = tid & 63;
  const int base = blockIdx.x * 1024;
  for (int r = (tid >> 6); r < 1024; r += 4) {
    int node = base + r;
    if (node >= N_) break;
    int g = batch[node];
    float v = h[(size_t)node * 64 + c];
    atomicAdd(&sg[g * 64 + c], v);
    if (c == 0) atomicAdd(&sc[g], 1.0f);
    if (with_bc) {
      float bc = xmask[node * 3 + 2];
      atomicAdd(&sb[g * 64 + c], v * bc);
      if (c == 0) atomicAdd(&sbc[g], bc);
    }
  }
  __syncthreads();
  for (int i = tid; i < G_ * H_; i += 256) {
    atomicAdd(&gsum[i], sg[i]);
    if (with_bc) atomicAdd(&bcsum[i], sb[i]);
  }
  if (tid < G_) {
    atomicAdd(&gcnt[tid], sc[tid]);
    if (with_bc) atomicAdd(&bccnt[tid], sbc[tid]);
  }
}

__global__ void k_norm(const float* __restrict__ gsum,
                       const float* __restrict__ gcnt, float* __restrict__ xg,
                       const float* __restrict__ bcsum,
                       const float* __restrict__ bccnt, float* __restrict__ xbc,
                       int with_bc) {
  const int i = threadIdx.x;  // 512 threads = G_*H_
  const int g = i >> 6;
  xg[i] = gsum[i] / fmaxf(gcnt[g], 1.0f);
  if (with_bc) xbc[i] = bcsum[i] / fmaxf(bccnt[g], 1.0f);
}

// ---------------------------------------------------------------------------
// Edge message MLP (131->64->64, K padded 160) fused with scatter-add.
// ---------------------------------------------------------------------------
__global__ __launch_bounds__(128) void k_msg(
    const float* __restrict__ h, const int* __restrict__ ei,
    const float* __restrict__ ea,
    const float* __restrict__ w1, const float* __restrict__ b1,
    const float* __restrict__ w2, const float* __restrict__ b2,
    float* __restrict__ agg, float* __restrict__ cnt) {
  __shared__ alignas(16) _Float16 sW1[5 * 4 * 512];   // 20 KB
  __shared__ alignas(16) _Float16 sW2[2 * 4 * 512];   // 8 KB
  __shared__ float sB1[64], sB2[64];
  __shared__ alignas(16) _Float16 sIn[4][16 * 160];   // 20 KB
  __shared__ int sSrc[4][16], sDst[4][16];
  const int tid = threadIdx.x, nth = blockDim.x;
  stage_Bsw(sW1, w1, 131, 64, 5, 4, tid, nth);
  stage_Bsw(sW2, w2, 64, 64, 2, 4, tid, nth);
  for (int i = tid; i < 64; i += nth) { sB1[i] = b1[i]; sB2[i] = b2[i]; }
  __syncthreads();
  const int wave = tid >> 5, lane = tid & 31;
  const int tile = blockIdx.x * 4 + wave;
  if (tile >= ET) return;
  const int e0 = tile * 16;
  if (lane < 16) {
    int e = e0 + lane;
    sSrc[wave][lane] = ei[e];
    int d = ei[E_ + e];
    sDst[wave][lane] = d;
    atomicAdd(&cnt[d], 1.0f);
  }
  _Float16* in = sIn[wave];
  for (int i = lane; i < 16 * 40; i += 32) {  // 40 groups of 4 per row
    int r = i / 40, g = i - r * 40;
    v4f v = {};
    if (g < 16)
      v = *(const v4f*)(h + (size_t)sSrc[wave][r] * 64 + g * 4);
    else if (g < 32)
      v = *(const v4f*)(h + (size_t)sDst[wave][r] * 64 + (g - 16) * 4);
    else if (g == 32) {
      int e = e0 + r;
      v[0] = ea[(size_t)e * 3 + 0];
      v[1] = ea[(size_t)e * 3 + 1];
      v[2] = ea[(size_t)e * 3 + 2];
    }
    *(v4h*)(in + r * 160 + g * 4) = cvt4(v);
  }
  v8f m[4];
  wave_mlp2<160, 5, 4>(in, sW1, sB1, sW2, sB2, in, lane, m);
  const int n0 = lane & 15, mh = (lane & 16) ? 8 : 0;
#pragma unroll
  for (int t = 0; t < 4; ++t)
#pragma unroll
    for (int r = 0; r < 8; ++r)
      atomicAdd(&agg[(size_t)sDst[wave][mh + r] * 64 + t * 16 + n0], m[t][r]);
}

// ---------------------------------------------------------------------------
// Node update MLP (256->64->64) with residual into h.
// ---------------------------------------------------------------------------
__global__ __launch_bounds__(64) void k_upd(
    float* __restrict__ h, const float* __restrict__ agg,
    const float* __restrict__ cnt, const int* __restrict__ batch,
    const float* __restrict__ xg, const float* __restrict__ xbc,
    const float* __restrict__ w1, const float* __restrict__ b1,
    const float* __restrict__ w2, const float* __restrict__ b2) {
  __shared__ alignas(16) _Float16 sW1[8 * 4 * 512];   // 32 KB
  __shared__ alignas(16) _Float16 sW2[2 * 4 * 512];   // 8 KB
  __shared__ float sB1[64], sB2[64];
  __shared__ alignas(16) _Float16 sIn[2][16 * 256];   // 16 KB
  const int tid = threadIdx.x, nth = blockDim.x;
  stage_Bsw(sW1, w1, 256, 64, 8, 4, tid, nth);
  stage_Bsw(sW2, w2, 64, 64, 2, 4, tid, nth);
  for (int i = tid; i < 64; i += nth) { sB1[i] = b1[i]; sB2[i] = b2[i]; }
  __syncthreads();
  const int wave = tid >> 5, lane = tid & 31;
  const int tile = blockIdx.x * 2 + wave;
  if (tile >= NT) return;
  const int row0 = tile * 16;
  _Float16* in = sIn[wave];
  for (int i = lane; i < 16 * 64; i += 32) {  // 64 groups of 4 per row
    int r = i >> 6, g = i & 63;
    int node = row0 + r;
    v4f v;
    if (g < 16) {
      v = *(const v4f*)(h + (size_t)node * 64 + g * 4);
    } else if (g < 32) {
      v = *(const v4f*)(agg + (size_t)node * 64 + (g - 16) * 4);
      float rc = 1.0f / fmaxf(cnt[node], 1.0f);
      v = v * rc;
    } else if (g < 48) {
      v = *(const v4f*)(xg + batch[node] * 64 + (g - 32) * 4);
    } else {
      v = *(const v4f*)(xbc + batch[node] * 64 + (g - 48) * 4);
    }
    *(v4h*)(in + r * 256 + g * 4) = cvt4(v);
  }
  v8f u[4];
  wave_mlp2<256, 8, 4>(in, sW1, sB1, sW2, sB2, in, lane, u);
  const int n0 = lane & 15, mh = (lane & 16) ? 8 : 0;
#pragma unroll
  for (int t = 0; t < 4; ++t)
#pragma unroll
    for (int r = 0; r < 8; ++r) {
      size_t idx = (size_t)(row0 + mh + r) * 64 + t * 16 + n0;
      h[idx] = h[idx] + u[t][r];
    }
}

// ---------------------------------------------------------------------------
// Decoder (64->64->4); layer-2 cols padded 4 -> 16 with zero weights.
// ---------------------------------------------------------------------------
__global__ __launch_bounds__(128) void k_dec(
    const float* __restrict__ h,
    const float* __restrict__ w1, const float* __restrict__ b1,
    const float* __restrict__ w2, const float* __restrict__ b2,
    float* __restrict__ out) {
  __shared__ alignas(16) _Float16 sW1[2 * 4 * 512];
  __shared__ alignas(16) _Float16 sW2[2 * 1 * 512];
  __shared__ float sB1[64], sB2[16];
  __shared__ alignas(16) _Float16 sIn[4][16 * 64];
  const int tid = threadIdx.x, nth = blockDim.x;
  stage_Bsw(sW1, w1, 64, 64, 2, 4, tid, nth);
  stage_Bsw(sW2, w2, 64, OUT_, 2, 1, tid, nth);
  for (int i = tid; i < 64; i += nth) sB1[i] = b1[i];
  if (tid < 16) sB2[tid] = (tid < OUT_) ? b2[tid] : 0.0f;
  __syncthreads();
  const int wave = tid >> 5, lane = tid & 31;
  const int tile = blockIdx.x * 4 + wave;
  if (tile >= NT) return;
  const int row0 = tile * 16;
  _Float16* in = sIn[wave];
  for (int i = lane; i < 16 * 16; i += 32) {
    int r = i >> 4, g = i & 15;
    v4f v = *(const v4f*)(h + (size_t)(row0 + r) * 64 + g * 4);
    *(v4h*)(in + r * 64 + g * 4) = cvt4(v);
  }
  v8f o[1];
  wave_mlp2<64, 2, 1>(in, sW1, sB1, sW2, sB2, in, lane, o);
  const int n0 = lane & 15, mh = (lane & 16) ? 8 : 0;
  if (n0 < OUT_) {
#pragma unroll
    for (int r = 0; r < 8; ++r)
      out[(size_t)(row0 + mh + r) * OUT_ + n0] = o[0][r];
  }
}

// ---------------------------------------------------------------------------
extern "C" void kernel_launch(void* const* d_in, const int* in_sizes, int n_in,
                              void* d_out, int out_size, void* d_ws,
                              size_t ws_size, hipStream_t stream) {
  (void)in_sizes; (void)n_in; (void)out_size; (void)ws_size;
  const float* x      = (const float*)d_in[0];
  const float* xmask  = (const float*)d_in[1];
  const int*   ei     = (const int*)d_in[2];
  const float* eattr  = (const float*)d_in[3];
  /* d_in[4] = pos (unused by reference output) */
  const int*   batch  = (const int*)d_in[5];
  const float* enc_w1 = (const float*)d_in[6];
  const float* enc_b1 = (const float*)d_in[7];
  const float* enc_w2 = (const float*)d_in[8];
  const float* enc_b2 = (const float*)d_in[9];
  const float* msg_w1 = (const float*)d_in[10];
  const float* msg_b1 = (const float*)d_in[11];
  const float* msg_w2 = (const float*)d_in[12];
  const float* msg_b2 = (const float*)d_in[13];
  const float* upd_w1 = (const float*)d_in[14];
  const float* upd_b1 = (const float*)d_in[15];
  const float* upd_w2 = (const float*)d_in[16];
  const float* upd_b2 = (const float*)d_in[17];
  const float* dec_w1 = (const float*)d_in[18];
  const float* dec_b1 = (const float*)d_in[19];
  const float* dec_w2 = (const float*)d_in[20];
  const float* dec_b2 = (const float*)d_in[21];

  float* h     = (float*)d_ws;                  // N*H
  float* agg   = h     + (size_t)N_ * H_;       // N*H
  float* cnt   = agg   + (size_t)N_ * H_;       // N
  float* gsum  = cnt   + N_;                    // G*H
  float* gcnt  = gsum  + G_ * H_;               // G
  float* bcsum = gcnt  + G_;                    // G*H
  float* bccnt = bcsum + G_ * H_;               // G
  float* xg    = bccnt + G_;                    // G*H
  float* xbc   = xg    + G_ * H_;               // G*H

  // Encoder
  k_encode<<<dim3((NT + 3) / 4), dim3(128), 0, stream>>>(
      x, xmask, enc_w1, enc_b1, enc_w2, enc_b2, h);

  // x_graph and x_BC (x_BC fixed for all rounds)
  hipMemsetAsync(gsum, 0, sizeof(float) * (2 * (G_ * H_ + G_)), stream);
  k_stats<<<dim3((N_ + 1023) / 1024), dim3(256), 0, stream>>>(
      h, batch, xmask, 1, gsum, gcnt, bcsum, bccnt);
  k_norm<<<dim3(1), dim3(512), 0, stream>>>(gsum, gcnt, xg, bcsum, bccnt, xbc, 1);

  for (int rep = 0; rep < 4; ++rep) {
    hipMemsetAsync(agg, 0, sizeof(float) * ((size_t)N_ * H_ + N_), stream);
    k_msg<<<dim3((ET + 3) / 4), dim3(128), 0, stream>>>(
        h, ei, eattr, msg_w1, msg_b1, msg_w2, msg_b2, agg, cnt);
    k_upd<<<dim3((NT + 1) / 2), dim3(64), 0, stream>>>(
        h, agg, cnt, batch, xg, xbc, upd_w1, upd_b1, upd_w2, upd_b2);
    if (rep < 3) {  // x_graph only consumed by next round's update
      hipMemsetAsync(gsum, 0, sizeof(float) * (G_ * H_ + G_), stream);
      k_stats<<<dim3((N_ + 1023) / 1024), dim3(256), 0, stream>>>(
          h, batch, xmask, 0, gsum, gcnt, bcsum, bccnt);
      k_norm<<<dim3(1), dim3(512), 0, stream>>>(gsum, gcnt, xg, bcsum, bccnt,
                                                xbc, 0);
    }
  }

  // Decoder
  k_dec<<<dim3((NT + 3) / 4), dim3(128), 0, stream>>>(
      h, dec_w1, dec_b1, dec_w2, dec_b2, (float*)d_out);
}
